// STGCN_43928925504131
// MI455X (gfx1250) — matrix operand order repaired and checked by reading, not compiled, not run on hardware
//
#include <hip/hip_runtime.h>
#include <hip/hip_bf16.h>
#include <math.h>

// ---------------------------------------------------------------------------
// STGCN forward for MI455X (gfx1250, wave32, WMMA 16x16x32 f16)
// B=16, N=1024, C_IN=2, T=12, HID=64, K_CHEB=3, KT=3, OUT*PRED=384, E=16384
// ---------------------------------------------------------------------------

typedef __attribute__((ext_vector_type(16))) _Float16 v16h;
typedef __attribute__((ext_vector_type(8)))  float    v8f;
typedef __attribute__((ext_vector_type(4)))  float    f4;

#define Bn   16
#define Nn   1024
#define Tt   12
#define HID  64
#define Ee   16384
#define T1   10          // T-2
#define T2T  8           // T-4
#define CO   384         // OUT*PRED
#define M1   (Bn*T1*Nn)  // 163840 rows of cheb GEMM
#define M2   (Bn*T2T*Nn) // 131072 rows of tc2 GEMM

__device__ __forceinline__ float sigmoidf_(float x) {
    return 1.0f / (1.0f + __expf(-x));
}

// load A fragment (contiguous k = 16*half + e, matching B) as 4x float4 + pack-convert
__device__ __forceinline__ v16h load_a_frag(const float* __restrict__ arow, int half) {
    const int o = half << 4;
    f4 a0 = *(const f4*)(arow + o);
    f4 a1 = *(const f4*)(arow + o + 4);
    f4 a2 = *(const f4*)(arow + o + 8);
    f4 a3 = *(const f4*)(arow + o + 12);
    v16h a;
    #pragma unroll
    for (int e = 0; e < 4; ++e) {
        a[e]      = (_Float16)a0[e];
        a[4 + e]  = (_Float16)a1[e];
        a[8 + e]  = (_Float16)a2[e];
        a[12 + e] = (_Float16)a3[e];
    }
    return a;
}

// ---------------------------------------------------------------------------
// Kernel 1: temporal conv 1  (B,N,C,T) -> T0 (B,T1,N,HID)
// ---------------------------------------------------------------------------
__global__ __launch_bounds__(256) void tc1_kernel(
    const float* __restrict__ X,
    const float* __restrict__ w1, const float* __restrict__ b1,
    const float* __restrict__ w2, const float* __restrict__ b2,
    const float* __restrict__ w3, const float* __restrict__ b3,
    float* __restrict__ T0)
{
    __shared__ float lw1[HID*6], lw2[HID*6], lw3[HID*6];
    __shared__ float lb1[HID], lb2[HID], lb3[HID];
    int tid = threadIdx.x;
    for (int i = tid; i < HID*6; i += 256) {
        lw1[i] = w1[i]; lw2[i] = w2[i]; lw3[i] = w3[i];
    }
    if (tid < HID) { lb1[tid] = b1[tid]; lb2[tid] = b2[tid]; lb3[tid] = b3[tid]; }
    __syncthreads();

    int idx = blockIdx.x * 256 + tid;          // over B*T1*N
    int n  = idx & (Nn-1);
    int bt = idx >> 10;
    int t  = bt % T1;
    int b  = bt / T1;

    float xv[2][3];
    #pragma unroll
    for (int c = 0; c < 2; ++c) {
        const float* xp = X + ((size_t)(b*Nn + n)*2 + c)*Tt + t;
        #pragma unroll
        for (int k = 0; k < 3; ++k) xv[c][k] = xp[k];
    }
    float* out = T0 + (size_t)idx * HID;
    #pragma unroll 4
    for (int h = 0; h < HID; ++h) {
        float P = lb1[h], Q = lb2[h], R = lb3[h];
        #pragma unroll
        for (int c = 0; c < 2; ++c) {
            #pragma unroll
            for (int k = 0; k < 3; ++k) {
                int wi = (h*2 + c)*3 + k;
                P += xv[c][k] * lw1[wi];
                Q += xv[c][k] * lw2[wi];
                R += xv[c][k] * lw3[wi];
            }
        }
        float v = P * sigmoidf_(Q) + R;
        out[h] = v > 0.f ? v : 0.f;
    }
}

// ---------------------------------------------------------------------------
// Kernel 2/3: graph normalization  deg = segsum(w, row); norm = -dinv_r*w*dinv_c
// ---------------------------------------------------------------------------
__global__ __launch_bounds__(256) void deg_kernel(
    const int* __restrict__ ei, const float* __restrict__ ew,
    float* __restrict__ deg)
{
    int e = blockIdx.x * 256 + threadIdx.x;
    if (e >= Ee) return;
    int r = ei[e], c = ei[Ee + e];
    float w = (r == c) ? 0.f : ew[e];
    atomicAdd(&deg[r], w);
}

__global__ __launch_bounds__(256) void norm_kernel(
    const int* __restrict__ ei, const float* __restrict__ ew,
    const float* __restrict__ deg, float* __restrict__ norm)
{
    int e = blockIdx.x * 256 + threadIdx.x;
    if (e >= Ee) return;
    int r = ei[e], c = ei[Ee + e];
    float w  = (r == c) ? 0.f : ew[e];
    float dr = deg[r], dc = deg[c];
    float ir = dr > 0.f ? rsqrtf(dr) : 0.f;
    float ic = dc > 0.f ? rsqrtf(dc) : 0.f;
    norm[e] = -(ir * w * ic);
}

// ---------------------------------------------------------------------------
// Kernel 4: sparse propagation  dst[bt, col, h] += src[bt, row, h] * norm[e]
// ---------------------------------------------------------------------------
__global__ __launch_bounds__(256) void prop_kernel(
    const float* __restrict__ src, float* __restrict__ dst,
    const int* __restrict__ ei, const float* __restrict__ norm)
{
    int idx = blockIdx.x * 256 + threadIdx.x;      // E * 160 * 64
    int e   = idx / (Bn*T1*HID);
    int rem = idx - e * (Bn*T1*HID);
    int bt  = rem >> 6;
    int h   = rem & 63;
    int r = ei[e], c = ei[Ee + e];
    float v = src[((size_t)bt*Nn + r)*HID + h] * norm[e];
    atomicAdd(&dst[((size_t)bt*Nn + c)*HID + h], v);
}

// Tx2 = 2*prop(Tx1) - Tx0
__global__ __launch_bounds__(256) void combine_kernel(
    const float* __restrict__ T0, float* __restrict__ Tx2)
{
    int idx = blockIdx.x * 256 + threadIdx.x;      // M1*HID
    Tx2[idx] = 2.f * Tx2[idx] - T0[idx];
}

// ---------------------------------------------------------------------------
// Kernel 5: Chebyshev GEMM (WMMA):
//   Tg(M1,64) = relu( [Tx0|Tx1|Tx2](M1,192) @ chebW(192,64) + bias )
// Block = 128 thr (4 waves); wave = 16-row strip x 64 cols.
// Weights pre-swizzled in LDS to per-(step,ntile,lane) contiguous fragments:
// the inner loop reads one 32-byte aligned v16h per fragment (ds_load_b128).
// ---------------------------------------------------------------------------
__global__ __launch_bounds__(128) void cheb_gemm(
    const float* __restrict__ Tx0, const float* __restrict__ Tx1,
    const float* __restrict__ Tx2,
    const float* __restrict__ W, const float* __restrict__ bias,
    float* __restrict__ Tg)
{
    // [step][ntile][lane][e] : 6*4*32*16 f16 = 24 KB
    __shared__ __align__(32) _Float16 ldsW[6*4*32*16];
    __shared__ float ldsB[64];
    int tid = threadIdx.x;
    for (int i = tid; i < 6*4*32*16; i += 128) {
        int e    = i & 15;
        int lane = (i >> 4) & 31;
        int nt   = (i >> 9) & 3;
        int s    = i >> 11;
        int k    = s*32 + ((lane >> 4) << 4) + e;   // B 32x16 fragment k-map
        int f    = nt*16 + (lane & 15);
        ldsW[i]  = (_Float16)W[k*64 + f];
    }
    if (tid < 64) ldsB[tid] = bias[tid];
    __syncthreads();

    int wave = tid >> 5, lane = tid & 31;
    int lane16 = lane & 15, half = lane >> 4;
    int mBase = blockIdx.x * 64 + wave * 16;
    int rowA  = mBase + lane16;
    const v16h* wfrag = (const v16h*)ldsW;

    v8f acc0 = {}, acc1 = {}, acc2 = {}, acc3 = {};

    #pragma unroll
    for (int step = 0; step < 6; ++step) {
        const int kBase = step * 32;
        const float* src = (kBase < 64) ? Tx0 : (kBase < 128) ? Tx1 : Tx2;
        const float* arow = src + (size_t)rowA * HID + (kBase & 63);
        v16h a = load_a_frag(arow, half);

        #pragma unroll
        for (int nt = 0; nt < 4; ++nt) {
            v16h bf = wfrag[(step*4 + nt)*32 + lane];
            if      (nt == 0) acc0 = __builtin_amdgcn_wmma_f32_16x16x32_f16(false, a, false, bf, (short)0, acc0, false, false);
            else if (nt == 1) acc1 = __builtin_amdgcn_wmma_f32_16x16x32_f16(false, a, false, bf, (short)0, acc1, false, false);
            else if (nt == 2) acc2 = __builtin_amdgcn_wmma_f32_16x16x32_f16(false, a, false, bf, (short)0, acc2, false, false);
            else              acc3 = __builtin_amdgcn_wmma_f32_16x16x32_f16(false, a, false, bf, (short)0, acc3, false, false);
        }
    }

    #pragma unroll
    for (int nt = 0; nt < 4; ++nt) {
        int f  = nt*16 + lane16;
        float bv = ldsB[f];
        #pragma unroll
        for (int r = 0; r < 8; ++r) {
            float v = ((nt==0)?acc0[r]:(nt==1)?acc1[r]:(nt==2)?acc2[r]:acc3[r]) + bv;
            v = v > 0.f ? v : 0.f;
            int gRow = mBase + r + (half<<3);
            Tg[(size_t)gRow * HID + f] = v;
        }
    }
}

// ---------------------------------------------------------------------------
// Kernel 6: temporal conv 2 as GEMM (WMMA), fused GLU epilogue.
//   A(M2,192): row(b,t,n), k = kt*64+h -> Tg[b, t+kt, n, h]
//   Bj(192,384): wj[(o*64+h)*3+kt]
//   H = relu(P * sigmoid(Q) + R)  -> T2 (B,8,N,384) f32
// Block = 128 thr; tile 64 rows x 32 cols x 3 convs. Weights fragment-swizzled.
// ---------------------------------------------------------------------------
__global__ __launch_bounds__(128) void tc2_gemm(
    const float* __restrict__ Tg,
    const float* __restrict__ w1, const float* __restrict__ b1,
    const float* __restrict__ w2, const float* __restrict__ b2,
    const float* __restrict__ w3, const float* __restrict__ b3,
    float* __restrict__ T2)
{
    // [conv][step][ntile][lane][e] : 3*6*2*32*16 f16 = 36 KB
    __shared__ __align__(32) _Float16 ldsW[3*6*2*32*16];
    __shared__ float ldsB[3][32];
    int tid = threadIdx.x;
    int oBase = blockIdx.y * 32;

    for (int i = tid; i < 3*6*2*32*16; i += 128) {
        int e    = i & 15;
        int lane = (i >> 4) & 31;
        int nt   = (i >> 9) & 1;
        int s    = (i >> 10) % 6;
        int j    = i / 6144;
        int k    = s*32 + ((lane >> 4) << 4) + e;
        int kt   = k >> 6, h = k & 63;
        int o    = oBase + nt*16 + (lane & 15);
        const float* wj = (j == 0) ? w1 : (j == 1) ? w2 : w3;
        ldsW[i] = (_Float16)wj[((size_t)o*HID + h)*3 + kt];
    }
    if (tid < 96) {
        int j = tid >> 5, ol = tid & 31;
        const float* bj = (j == 0) ? b1 : (j == 1) ? b2 : b3;
        ldsB[j][ol] = bj[oBase + ol];
    }
    __syncthreads();

    int wave = tid >> 5, lane = tid & 31;
    int lane16 = lane & 15, half = lane >> 4;
    int mBase = blockIdx.x * 64 + wave * 16;
    int mRow  = mBase + lane16;
    int n   = mRow & (Nn-1);
    int bt8 = mRow >> 10;
    int t   = bt8 & 7;
    int b   = bt8 >> 3;
    const v16h* wfrag = (const v16h*)ldsW;

    v8f p0 = {}, p1 = {}, q0 = {}, q1 = {}, r0 = {}, r1 = {};

    #pragma unroll
    for (int step = 0; step < 6; ++step) {
        const int kBase = step * 32;
        const int kt = kBase >> 6;
        const float* arow = Tg + ((size_t)((b*T1 + t + kt)*Nn + n))*HID + (kBase & 63);
        v16h a = load_a_frag(arow, half);

        #pragma unroll
        for (int nt = 0; nt < 2; ++nt) {
            // conv j fragment index: ((j*6 + step)*2 + nt)*32 + lane
            v16h bP = wfrag[((0*6 + step)*2 + nt)*32 + lane];
            v16h bQ = wfrag[((1*6 + step)*2 + nt)*32 + lane];
            v16h bR = wfrag[((2*6 + step)*2 + nt)*32 + lane];
            if (nt == 0) {
                p0 = __builtin_amdgcn_wmma_f32_16x16x32_f16(false, a, false, bP, (short)0, p0, false, false);
                q0 = __builtin_amdgcn_wmma_f32_16x16x32_f16(false, a, false, bQ, (short)0, q0, false, false);
                r0 = __builtin_amdgcn_wmma_f32_16x16x32_f16(false, a, false, bR, (short)0, r0, false, false);
            } else {
                p1 = __builtin_amdgcn_wmma_f32_16x16x32_f16(false, a, false, bP, (short)0, p1, false, false);
                q1 = __builtin_amdgcn_wmma_f32_16x16x32_f16(false, a, false, bQ, (short)0, q1, false, false);
                r1 = __builtin_amdgcn_wmma_f32_16x16x32_f16(false, a, false, bR, (short)0, r1, false, false);
            }
        }
    }

    #pragma unroll
    for (int nt = 0; nt < 2; ++nt) {
        int col = nt*16 + lane16;
        int o   = oBase + col;
        float bP = ldsB[0][col], bQ = ldsB[1][col], bR = ldsB[2][col];
        #pragma unroll
        for (int r = 0; r < 8; ++r) {
            float P = ((nt==0)?p0[r]:p1[r]) + bP;
            float Q = ((nt==0)?q0[r]:q1[r]) + bQ;
            float R = ((nt==0)?r0[r]:r1[r]) + bR;
            float H = P * sigmoidf_(Q) + R;
            H = H > 0.f ? H : 0.f;
            int gRow = mBase + r + (half<<3);
            T2[(size_t)gRow * CO + o] = H;
        }
    }
}

// ---------------------------------------------------------------------------
// Kernel 7: per-node BN statistics over (B, T'', Cout) = 49152 elems/node
// ---------------------------------------------------------------------------
__global__ __launch_bounds__(256) void bn_stats(
    const float* __restrict__ T2, float* __restrict__ mean, float* __restrict__ rvar)
{
    int n = blockIdx.x, tid = threadIdx.x;
    float s = 0.f, ss = 0.f;
    for (int i = tid; i < Bn*T2T*CO; i += 256) {
        int bt = i / CO;
        int c  = i - bt * CO;
        float v = T2[((size_t)bt*Nn + n)*CO + c];
        s += v; ss += v * v;
    }
    __shared__ float sh[256], sh2[256];
    sh[tid] = s; sh2[tid] = ss;
    __syncthreads();
    for (int off = 128; off > 0; off >>= 1) {
        if (tid < off) { sh[tid] += sh[tid+off]; sh2[tid] += sh2[tid+off]; }
        __syncthreads();
    }
    if (tid == 0) {
        const float inv = 1.f / (float)(Bn*T2T*CO);
        float m = sh[0] * inv;
        float v = sh2[0] * inv - m*m;
        mean[n] = m;
        rvar[n] = rsqrtf(v + 1e-5f);
    }
}

// ---------------------------------------------------------------------------
// Kernel 8: output = gamma[n]*(T2[b,7,n,c]-mean)*rvar + beta, flat (c,b,n)
// ---------------------------------------------------------------------------
__global__ __launch_bounds__(256) void bn_out(
    const float* __restrict__ T2,
    const float* __restrict__ mean, const float* __restrict__ rvar,
    const float* __restrict__ gamma, const float* __restrict__ beta,
    float* __restrict__ out)
{
    int idx = blockIdx.x * 256 + threadIdx.x;   // 384*16*1024
    int c   = idx >> 14;
    int rem = idx & 16383;
    int b   = rem >> 10;
    int n   = rem & 1023;
    float v = T2[((size_t)(b*T2T + 7)*Nn + n)*CO + c];
    out[idx] = gamma[n] * (v - mean[n]) * rvar[n] + beta[n];
}

// ---------------------------------------------------------------------------
extern "C" void kernel_launch(void* const* d_in, const int* in_sizes, int n_in,
                              void* d_out, int out_size, void* d_ws, size_t ws_size,
                              hipStream_t stream) {
    const float* X   = (const float*)d_in[0];
    const int*   ei  = (const int*)  d_in[1];
    const float* ew  = (const float*)d_in[2];
    const float* t1w1 = (const float*)d_in[3];  const float* t1b1 = (const float*)d_in[4];
    const float* t1w2 = (const float*)d_in[5];  const float* t1b2 = (const float*)d_in[6];
    const float* t1w3 = (const float*)d_in[7];  const float* t1b3 = (const float*)d_in[8];
    const float* chW  = (const float*)d_in[9];  const float* chB  = (const float*)d_in[10];
    const float* t2w1 = (const float*)d_in[11]; const float* t2b1 = (const float*)d_in[12];
    const float* t2w2 = (const float*)d_in[13]; const float* t2b2 = (const float*)d_in[14];
    const float* t2w3 = (const float*)d_in[15]; const float* t2b3 = (const float*)d_in[16];
    const float* gma  = (const float*)d_in[17]; const float* bta  = (const float*)d_in[18];
    float* out = (float*)d_out;

    // workspace layout (floats)
    const size_t SEG = (size_t)M1 * HID;        // 10,485,760
    float* ws   = (float*)d_ws;
    float* deg  = ws;                            // 1024
    float* norm = ws + 2048;                     // 16384
    float* mean = ws + 2048 + Ee;                // 1024
    float* rvar = mean + 1024;                   // 1024
    float* Tg   = ws + 262144;                   // SEG
    float* T0   = Tg  + SEG;                     // SEG
    float* Tx1  = T0  + SEG;                     // SEG
    float* Tx2  = Tx1 + SEG;                     // SEG
    float* T2b  = T0;                            // reuse (T0/Tx1/Tx2 dead by then), 50,331,648

    hipMemsetAsync(deg, 0, Nn * sizeof(float), stream);
    hipMemsetAsync(Tx1, 0, SEG * sizeof(float), stream);
    hipMemsetAsync(Tx2, 0, SEG * sizeof(float), stream);

    // 1) temporal conv 1
    tc1_kernel<<<M1/256, 256, 0, stream>>>(X, t1w1, t1b1, t1w2, t1b2, t1w3, t1b3, T0);

    // 2) graph normalization
    deg_kernel <<<Ee/256, 256, 0, stream>>>(ei, ew, deg);
    norm_kernel<<<Ee/256, 256, 0, stream>>>(ei, ew, deg, norm);

    // 3) Chebyshev propagations
    const int propThreads = Ee * Bn * T1 * HID;  // 167,772,160
    prop_kernel<<<propThreads/256, 256, 0, stream>>>(T0,  Tx1, ei, norm);
    prop_kernel<<<propThreads/256, 256, 0, stream>>>(Tx1, Tx2, ei, norm);
    combine_kernel<<<(int)(SEG/256), 256, 0, stream>>>(T0, Tx2);

    // 4) Chebyshev GEMM (WMMA) + relu -> Tg
    cheb_gemm<<<M1/64, 128, 0, stream>>>(T0, Tx1, Tx2, chW, chB, Tg);

    // 5) temporal conv 2 GEMM (WMMA) + GLU -> T2
    dim3 g2(M2/64, CO/32);
    tc2_gemm<<<g2, 128, 0, stream>>>(Tg, t2w1, t2b1, t2w2, t2b2, t2w3, t2b3, T2b);

    // 6) batch norm over node channel + final gather
    bn_stats<<<Nn, 256, 0, stream>>>(T2b, mean, rvar);
    bn_out<<<(CO*Bn*Nn)/256, 256, 0, stream>>>(T2b, mean, rvar, gma, bta, out);
}
